// SequenceEncoder_1297080123897
// MI455X (gfx1250) — matrix operand order repaired and checked by reading, compiled, not run on hardware
//
#include <hip/hip_runtime.h>
#include <hip/hip_bf16.h>

#define NUM_SEG 512
#define MAX_LEN 256
#define DIM 256
#define LAT 64

typedef __attribute__((ext_vector_type(16))) _Float16 v16h;
typedef __attribute__((ext_vector_type(8)))  float    v8f;
typedef __attribute__((ext_vector_type(4))) unsigned int u32x4;
typedef __attribute__((ext_vector_type(8))) int i32x8;
typedef __attribute__((ext_vector_type(4))) int i32x4;

__device__ __forceinline__ v8f zero8() {
    v8f z;
#pragma unroll
    for (int i = 0; i < 8; ++i) z[i] = 0.0f;
    return z;
}

__device__ __forceinline__ v8f wmma_f16(v16h a, v16h b, v8f c) {
    // D = A(16x32 f16) * B(32x16 f16) + C(16x16 f32)
    return __builtin_amdgcn_wmma_f32_16x16x32_f16(
        /*neg_a=*/false, a, /*neg_b=*/false, b,
        /*c_mod=*/(short)0, c, /*reuse_a=*/false, /*reuse_b=*/false);
}

// A fragment: 16x32 f16 tile, row-major source, base points at (row0, k0).
// Layout (this code's contiguous k order; B loaders use the same): lane m=lane&15, group g=lane>>4,
// element j -> k = 16*g + j   (32B per lane)
__device__ __forceinline__ v16h load_a_frag(const _Float16* base, int ld, int lane) {
    int m = lane & 15;
    int g = lane >> 4;
    v16h a;
#pragma unroll
    for (int j = 0; j < 16; ++j) {
        int k = (g << 4) + j;
        a[j] = base[m * ld + k];
    }
    return a;
}

// B fragment where memory holds the TRANSPOSE of B: element B(k,n) = base[n*ld + k].
// B layout: lane n=lane&15, group g=lane>>4, element j -> k = 16*g + j (32B per lane)
__device__ __forceinline__ v16h load_bt_frag(const _Float16* base, int ld, int lane) {
    int n = lane & 15;
    int g = lane >> 4;
    v16h b;
#pragma unroll
    for (int j = 0; j < 16; ++j) {
        b[j] = base[n * ld + (g << 4) + j];
    }
    return b;
}

// ---------------------------------------------------------------------------
// TDM: async 2D f16 tile load Global -> LDS via Tensor Data Mover.
// D# built per ISA 8.3/8.4: group0 {count=1, lds_addr, global_addr, type=2},
// group1 {data_size=1 (2B), tensor_dim0/1, tile_dim0/1, dim0_stride}.
// Tracked on TENSORcnt; caller waits with s_wait_tensorcnt then barriers.
// This toolchain exposes the 6-arg builtin:
//   (u32x4 g0, i32x8 g1, i32x4 g2, i32x4 g3, i32x8 g4, i32 cpol)
// ---------------------------------------------------------------------------
#if __has_builtin(__builtin_amdgcn_tensor_load_to_lds)
#define HAVE_TDM 1
__device__ __forceinline__ void tdm_load_2d_f16(unsigned lds_byte_off,
                                                const void* gptr,
                                                unsigned dim0, unsigned dim1,
                                                unsigned stride0)
{
    unsigned long long ga = (unsigned long long)(uintptr_t)gptr;
    u32x4 g0;
    i32x8 g1, g4;
    i32x4 g2, g3;
#pragma unroll
    for (int i = 0; i < 8; ++i) { g1[i] = 0; g4[i] = 0; }
#pragma unroll
    for (int i = 0; i < 4; ++i) { g2[i] = 0; g3[i] = 0; }
    g0[0] = 1u;                                            // count=1 (valid), user mode
    g0[1] = lds_byte_off;                                  // lds_addr [63:32]
    g0[2] = (unsigned)(ga & 0xffffffffu);                  // global_addr [95:64]
    g0[3] = (unsigned)((ga >> 32) & 0x01ffffffu)           // global_addr [120:96]
          | (2u << 30);                                    // type=2 ("image")
    g1[0] = (int)(1u << 16);                               // wg_mask=0, data_size=1 (2B)
    g1[1] = (int)((dim0 & 0xffffu) << 16);                 // tensor_dim0 [63:48]
    g1[2] = (int)((dim0 >> 16) | ((dim1 & 0xffffu) << 16)); // dim0 hi | tensor_dim1 lo
    g1[3] = (int)((dim1 >> 16) | ((dim0 & 0xffffu) << 16)); // dim1 hi | tile_dim0
    g1[4] = (int)(dim1 & 0xffffu);                         // tile_dim1 (tile_dim2=0)
    g1[5] = (int)stride0;                                  // tensor_dim0_stride lo32
    __builtin_amdgcn_tensor_load_to_lds(g0, g1, g2, g3, g4, 0);
}
#else
#define HAVE_TDM 0
#endif

// ---------------------------------------------------------------------------
// Kernel 1: masked gather + Q/K/V projections (f16 out), WMMA 16x16x32
// grid (NUM_SEG, 4 row-tiles of 64), block 256 (8 waves)
// LDS: X tile 64x256 f16 (32 KB) + W 256x256 f16 (128 KB) = 160 KB dynamic
// ---------------------------------------------------------------------------
__global__ void qkv_kernel(const float* __restrict__ seq,
                           const float* __restrict__ wq, const float* __restrict__ bq,
                           const float* __restrict__ wk, const float* __restrict__ bk,
                           const float* __restrict__ wv, const float* __restrict__ bv,
                           const long long* __restrict__ sse,
                           _Float16* __restrict__ Qo, _Float16* __restrict__ Ko,
                           _Float16* __restrict__ Vo, int max_row)
{
    extern __shared__ char smem_raw[];
    _Float16* Xs = (_Float16*)smem_raw;       // 64*256
    _Float16* Ws = Xs + 64 * DIM;             // 256*256

    const int b    = blockIdx.x;
    const int rt   = blockIdx.y;
    const int tid  = threadIdx.x;
    const int lane = tid & 31;
    const int wave = tid >> 5;

    const long long start = sse[2 * b];
    const int len  = (int)(sse[2 * b + 1] - start);
    const int row0 = rt * 64;

    // gather + mask 64 rows into LDS as f16 (4 threads per row, 64 cols each)
    {
        int r  = tid >> 2;
        int c0 = (tid & 3) * 64;
        int l  = row0 + r;
        bool valid = l < len;
        long long g = start + l;
        if (g > max_row) g = (long long)max_row;
        const float* src = seq + g * DIM + c0;
#pragma unroll 8
        for (int c = 0; c < 64; ++c)
            Xs[r * DIM + c0 + c] = valid ? (_Float16)src[c] : (_Float16)0.0f;
    }

    const float* Wm[3]  = { wq, wk, wv };
    const float* Bm[3]  = { bq, bk, bv };
    _Float16*    Om[3]  = { Qo, Ko, Vo };

#pragma unroll
    for (int mat = 0; mat < 3; ++mat) {
        __syncthreads();
        const float* W = Wm[mat];
        for (int i = tid; i < DIM * DIM; i += 256)
            Ws[i] = (_Float16)W[i];             // row-major [out][in]
        __syncthreads();

        // wave handles N-tiles {2w, 2w+1}; 4 M-tiles each
        v8f acc[4][2];
#pragma unroll
        for (int m = 0; m < 4; ++m)
#pragma unroll
            for (int n = 0; n < 2; ++n) acc[m][n] = zero8();

        for (int k0 = 0; k0 < DIM; k0 += 32) {
            v16h a[4], bf[2];
#pragma unroll
            for (int m = 0; m < 4; ++m)
                a[m] = load_a_frag(Xs + (m * 16) * DIM + k0, DIM, lane);
#pragma unroll
            for (int n = 0; n < 2; ++n)
                bf[n] = load_bt_frag(Ws + ((wave * 2 + n) * 16) * DIM + k0, DIM, lane);
#pragma unroll
            for (int m = 0; m < 4; ++m)
#pragma unroll
                for (int n = 0; n < 2; ++n)
                    acc[m][n] = wmma_f16(a[m], bf[n], acc[m][n]);
        }

        _Float16* O = Om[mat] + (size_t)b * DIM * DIM;
        const float* bias = Bm[mat];
        const int g2 = lane >> 4, c = lane & 15;
#pragma unroll
        for (int n = 0; n < 2; ++n) {
            int col = (wave * 2 + n) * 16 + c;
            float bb = bias[col];
#pragma unroll
            for (int m = 0; m < 4; ++m)
#pragma unroll
                for (int r = 0; r < 8; ++r) {
                    int row = row0 + m * 16 + r + g2 * 8;
                    O[row * DIM + col] = (_Float16)(acc[m][n][r] + bb);
                }
        }
    }
}

// ---------------------------------------------------------------------------
// Kernel 2: K tile -> LDS via TDM; scores = QK^T (WMMA) -> masked softmax
// (in-register, xor-shuffles) -> column-sum over valid rows -> pooled =
// colsum @ V / len.  grid NUM_SEG, block 256 (8 waves). LDS: K 128 KB + 1 KB.
// ---------------------------------------------------------------------------
__global__ void attn_pool_kernel(const _Float16* __restrict__ Q,
                                 const _Float16* __restrict__ K,
                                 const _Float16* __restrict__ V,
                                 const long long* __restrict__ sse,
                                 _Float16* __restrict__ pooled)
{
    extern __shared__ char smem_raw[];
    _Float16* Ks    = (_Float16*)smem_raw;              // 256*256
    float*    csum  = (float*)(Ks + DIM * DIM);         // 256

    const int b = blockIdx.x, tid = threadIdx.x;
    const int lane = tid & 31, wave = tid >> 5;
    const long long start = sse[2 * b];
    const int len = (int)(sse[2 * b + 1] - start);

    const _Float16* Kg = K + (size_t)b * DIM * DIM;
    const _Float16* Qg = Q + (size_t)b * DIM * DIM;
    const _Float16* Vg = V + (size_t)b * DIM * DIM;

#if HAVE_TDM
    if (wave == 0) {
        // async DMA of the full 256x256 f16 K tile into LDS (TENSORcnt)
        tdm_load_2d_f16((unsigned)(uintptr_t)Ks, Kg, DIM, DIM, DIM);
#if __has_builtin(__builtin_amdgcn_s_wait_tensorcnt)
        __builtin_amdgcn_s_wait_tensorcnt(0);
#endif
    }
#else
    for (int i = tid; i < DIM * DIM; i += 256) Ks[i] = Kg[i];
#endif
    if (tid < DIM) csum[tid] = 0.0f;
    // warm V into cache while we do attention (global_prefetch_b8)
    __builtin_prefetch(Vg + tid * DIM, 0, 0);
    __syncthreads();   // releases waves 1..7 after TDM data has landed

    const float scale = 0.0625f;  // 1/sqrt(256)
    const int g = lane >> 4, c = lane & 15;

    for (int mt = wave; mt < 16; mt += 8) {  // 16-row query strips
        v8f acc[16];
#pragma unroll
        for (int t = 0; t < 16; ++t) acc[t] = zero8();

        for (int k0 = 0; k0 < DIM; k0 += 32) {
            v16h a = load_a_frag(Qg + (mt * 16) * DIM + k0, DIM, lane);
#pragma unroll
            for (int t = 0; t < 16; ++t) {
                v16h bf = load_bt_frag(Ks + (t * 16) * DIM + k0, DIM, lane);
                acc[t] = wmma_f16(a, bf, acc[t]);
            }
        }

        // scale + key mask (col >= len -> -1e9)
#pragma unroll
        for (int t = 0; t < 16; ++t) {
            int col = t * 16 + c;
            bool bad = col >= len;
#pragma unroll
            for (int r = 0; r < 8; ++r)
                acc[t][r] = bad ? -1.0e9f : acc[t][r] * scale;
        }

        // row softmax: each row's 256 values live in one 16-lane group,
        // spread over 16 tile-registers -> register max/sum + xor-shuffles 8..1
#pragma unroll
        for (int r = 0; r < 8; ++r) {
            float mx = -3.4e38f;
#pragma unroll
            for (int t = 0; t < 16; ++t) mx = fmaxf(mx, acc[t][r]);
#pragma unroll
            for (int o = 8; o >= 1; o >>= 1) mx = fmaxf(mx, __shfl_xor(mx, o, 32));
            float sm = 0.0f;
#pragma unroll
            for (int t = 0; t < 16; ++t) {
                float e = __expf(acc[t][r] - mx);
                acc[t][r] = e;
                sm += e;
            }
#pragma unroll
            for (int o = 8; o >= 1; o >>= 1) sm += __shfl_xor(sm, o, 32);
            float inv = 1.0f / sm;
#pragma unroll
            for (int t = 0; t < 16; ++t) acc[t][r] *= inv;
        }

        // column-sum over VALID query rows only (pooling fuses attn@V)
#pragma unroll
        for (int t = 0; t < 16; ++t) {
            float s = 0.0f;
#pragma unroll
            for (int r = 0; r < 8; ++r) {
                int row = mt * 16 + r + g * 8;
                s += (row < len) ? acc[t][r] : 0.0f;
            }
            s += __shfl_xor(s, 16, 32);  // merge the two row half-groups
            if (lane < 16) atomicAdd(&csum[t * 16 + c], s);
        }
    }
    __syncthreads();

    // pooled[d] = (colsum . V[:,d]) / len   (coalesced V reads across threads)
    int d = tid;
    float p = 0.0f;
#pragma unroll 4
    for (int m = 0; m < DIM; ++m)
        p += csum[m] * (float)Vg[m * DIM + d];
    pooled[(size_t)b * DIM + d] = (_Float16)(p / (float)len);
}

// ---------------------------------------------------------------------------
// Kernel 3: mean / log_var heads: [512x256] @ [256x64]^T via WMMA
// grid 8 (64 rows each), block 256: waves 0-3 -> mean N-tiles, 4-7 -> lv
// ---------------------------------------------------------------------------
__global__ void head_kernel(const _Float16* __restrict__ pooled,
                            const float* __restrict__ wm, const float* __restrict__ bm,
                            const float* __restrict__ wl, const float* __restrict__ bl,
                            float* __restrict__ out)
{
    extern __shared__ char smem_raw[];
    _Float16* Ps = (_Float16*)smem_raw;   // 64*256
    const int tid = threadIdx.x, lane = tid & 31, wave = tid >> 5;
    const int row0 = blockIdx.x * 64;

    for (int i = tid; i < 64 * DIM; i += 256)
        Ps[i] = pooled[(size_t)row0 * DIM + i];
    __syncthreads();

    const int matsel = wave >> 2;     // 0 = mean, 1 = log_var
    const int nt     = wave & 3;      // N tile within the 64 outputs
    const float* W    = matsel ? wl : wm;
    const float* bias = matsel ? bl : bm;
    float* O = out + (size_t)matsel * NUM_SEG * LAT;

    const int g = lane >> 4, c = lane & 15;
    v8f acc[4];
#pragma unroll
    for (int m = 0; m < 4; ++m) acc[m] = zero8();

    for (int k0 = 0; k0 < DIM; k0 += 32) {
        v16h bf;   // B(k,n) = W[(nt*16+n)*256 + k0+k], on-the-fly f32->f16
#pragma unroll
        for (int j = 0; j < 16; ++j)
            bf[j] = (_Float16)W[(nt * 16 + c) * DIM + k0 + (g << 4) + j];
#pragma unroll
        for (int m = 0; m < 4; ++m) {
            v16h a = load_a_frag(Ps + (m * 16) * DIM + k0, DIM, lane);
            acc[m] = wmma_f16(a, bf, acc[m]);
        }
    }

    float bb = bias[nt * 16 + c];
#pragma unroll
    for (int m = 0; m < 4; ++m)
#pragma unroll
        for (int r = 0; r < 8; ++r) {
            int row = row0 + m * 16 + r + g * 8;
            O[(size_t)row * LAT + nt * 16 + c] = acc[m][r] + bb;
        }
}

// ---------------------------------------------------------------------------
extern "C" void kernel_launch(void* const* d_in, const int* in_sizes, int n_in,
                              void* d_out, int out_size, void* d_ws, size_t ws_size,
                              hipStream_t stream)
{
    const float*     seq = (const float*)d_in[0];
    const float*     wq  = (const float*)d_in[1];
    const float*     bq  = (const float*)d_in[2];
    const float*     wk  = (const float*)d_in[3];
    const float*     bk  = (const float*)d_in[4];
    const float*     wv  = (const float*)d_in[5];
    const float*     bv  = (const float*)d_in[6];
    const float*     wm  = (const float*)d_in[7];
    const float*     bm  = (const float*)d_in[8];
    const float*     wl  = (const float*)d_in[9];
    const float*     bl  = (const float*)d_in[10];
    const long long* sse = (const long long*)d_in[11];

    const int max_row = in_sizes[0] / DIM - 1;

    // workspace layout (f16): Q, K, V [512*256*256] + pooled [512*256]
    const size_t mat_elems = (size_t)NUM_SEG * DIM * DIM;
    _Float16* Q      = (_Float16*)d_ws;
    _Float16* K      = Q + mat_elems;
    _Float16* V      = K + mat_elems;
    _Float16* pooled = V + mat_elems;

    const size_t lds1 = (64 * DIM + DIM * DIM) * sizeof(_Float16);          // 160 KB
    const size_t lds2 = DIM * DIM * sizeof(_Float16) + DIM * sizeof(float); // 129 KB
    const size_t lds3 = 64 * DIM * sizeof(_Float16);                        // 32 KB

    qkv_kernel<<<dim3(NUM_SEG, MAX_LEN / 64), 256, lds1, stream>>>(
        seq, wq, bq, wk, bk, wv, bv, sse, Q, K, V, max_row);

    attn_pool_kernel<<<NUM_SEG, 256, lds2, stream>>>(Q, K, V, sse, pooled);

    head_kernel<<<MAX_LEN * 2 / 64, 256, lds3, stream>>>(
        pooled, wm, bm, wl, bl, (float*)d_out);
}
